// PointNet2ClsGeoerror_56839597195596
// MI455X (gfx1250) — matrix operand order repaired and checked by reading, not compiled, not run on hardware
//
#include <hip/hip_runtime.h>
#include <stdint.h>

typedef _Float16 half_t;
typedef __attribute__((ext_vector_type(16))) _Float16 v16h;
typedef __attribute__((ext_vector_type(8)))  float    v8f;

// ------------------------------------------------- CDNA5 async global->LDS
// GLOBAL_LOAD_ASYNC_TO_LDS_B128 (opcode 96..98 family): VDST = LDS byte
// address VGPR, VADDR = 64-bit global address. Tracked by ASYNCcnt.
__device__ __forceinline__ void async_copy16(uint32_t lds_off, const void* gptr) {
  asm volatile("global_load_async_to_lds_b128 %0, %1, off"
               :: "v"(lds_off), "v"((uint64_t)(uintptr_t)gptr)
               : "memory");
}
__device__ __forceinline__ void wait_async() {
  asm volatile("s_wait_asynccnt 0x0" ::: "memory");
}

// ---------------------------------------------------------------- utilities

__global__ void zero_kernel(float* __restrict__ p, int n) {
  int i = blockIdx.x * blockDim.x + threadIdx.x;
  if (i < n) p[i] = 0.f;
}

// xyz (B,C,N) -> pts (B,N,C), C=4
__global__ void transpose_xyz_kernel(const float* __restrict__ in, float* __restrict__ out,
                                     int B, int C, int N) {
  int i = blockIdx.x * blockDim.x + threadIdx.x;
  int total = B * C * N;
  if (i >= total) return;
  int b = i / (C * N);
  int r = i % (C * N);
  int c = r / N;
  int n = r % N;
  out[((size_t)b * N + n) * C + c] = in[i];
}

// W (Nout x K) f32 -> Wh (rows x Kpad) f16, K zero-padded to Kpad
__global__ void convert_w_kernel(const float* __restrict__ W, int Nout, int K, int Kpad,
                                 half_t* __restrict__ Wh) {
  int i = blockIdx.x * blockDim.x + threadIdx.x;
  if (i >= Nout * Kpad) return;
  int n = i / Kpad, k = i % Kpad;
  Wh[i] = (half_t)((k < K) ? W[(size_t)n * K + k] : 0.f);
}

// in-place Y = relu(scale[c]*Y + shift[c]); folded batch-norm application
__global__ void bnrelu_kernel(half_t* __restrict__ Y, const float* __restrict__ sc,
                              const float* __restrict__ sh, int M, int C) {
  int i = blockIdx.x * blockDim.x + threadIdx.x;
  if (i >= M * C) return;
  int c = i % C;
  float v = (float)Y[i] * sc[c] + sh[c];
  Y[i] = (half_t)(v > 0.f ? v : 0.f);
}

// ---------------------------------------------------------------- FPS
__global__ void fps_kernel(const float* __restrict__ pts, int N, int npoint,
                           int* __restrict__ idx_out) {
  __shared__ float dist_s[4096];
  __shared__ float wmax[8];
  __shared__ int   wimax[8];
  __shared__ int   far_s;
  int b = blockIdx.x;
  const float* Pb = pts + (size_t)b * N * 4;
  int tid = threadIdx.x, lane = tid & 31, wave = tid >> 5;

  for (int i = tid; i < N; i += 256) dist_s[i] = 1e10f;
  if (tid == 0) far_s = 0;
  __syncthreads();

  for (int it = 0; it < npoint; ++it) {
    int far = far_s;
    if (tid == 0) idx_out[b * npoint + it] = far;
    float cx = Pb[far * 4 + 0], cy = Pb[far * 4 + 1];
    float cz = Pb[far * 4 + 2], cw = Pb[far * 4 + 3];
    float best = -1.f; int besti = 0;
    for (int i = tid; i < N; i += 256) {
      float dx = Pb[i * 4 + 0] - cx, dy = Pb[i * 4 + 1] - cy;
      float dz = Pb[i * 4 + 2] - cz, dw = Pb[i * 4 + 3] - cw;
      float d = dx * dx + dy * dy + dz * dz + dw * dw;
      float dm = dist_s[i];
      if (d < dm) { dm = d; dist_s[i] = d; }
      if (dm > best || (dm == best && i < besti)) { best = dm; besti = i; }
    }
    for (int off = 16; off > 0; off >>= 1) {
      float ob = __shfl_down(best, off, 32);
      int   oi = __shfl_down(besti, off, 32);
      if (ob > best || (ob == best && oi < besti)) { best = ob; besti = oi; }
    }
    if (lane == 0) { wmax[wave] = best; wimax[wave] = besti; }
    __syncthreads();
    if (tid == 0) {
      float bb = wmax[0]; int bi = wimax[0];
      for (int w = 1; w < 8; ++w)
        if (wmax[w] > bb || (wmax[w] == bb && wimax[w] < bi)) { bb = wmax[w]; bi = wimax[w]; }
      far_s = bi;
    }
    __syncthreads();
  }
}

__global__ void gather_xyz_kernel(const float* __restrict__ pts, const int* __restrict__ idx,
                                  int N, int S, int total, float* __restrict__ out) {
  int i = blockIdx.x * blockDim.x + threadIdx.x;
  if (i >= total) return;
  int b = i / (S * 4);
  int r = i % (S * 4);
  int s = r >> 2, c = r & 3;
  out[i] = pts[((size_t)b * N + idx[b * S + s]) * 4 + c];
}

// ---------------------------------------------------------------- ball query + grouping
// One wave32 per query point; writes grouped rows with stride outCpad
// (zero-padded so the GEMM can async-copy K in 32-wide chunks).
__global__ void ballquery_group_kernel(const float* __restrict__ pts,
                                       const half_t* __restrict__ feat,
                                       const float* __restrict__ new_xyz,
                                       int B, int N, int S, int nsample, float r2,
                                       int featC, int outCpad, half_t* __restrict__ out) {
  __shared__ int idx_s[8][72];
  int tid = threadIdx.x, lane = tid & 31, wv = tid >> 5;
  int gw = blockIdx.x * 8 + wv;
  int BS = B * S;
  bool valid = gw < BS;
  int b = valid ? (gw / S) : 0;
  const float* Pb = pts + (size_t)b * N * 4;
  float cx = 0.f, cy = 0.f, cz = 0.f, cw = 0.f;
  if (valid) {
    cx = new_xyz[(size_t)gw * 4 + 0]; cy = new_xyz[(size_t)gw * 4 + 1];
    cz = new_xyz[(size_t)gw * 4 + 2]; cw = new_xyz[(size_t)gw * 4 + 3];
  }
  int total = 0;
  if (valid) {
    for (int j0 = 0; j0 < N && total < nsample; j0 += 32) {
      int j = j0 + lane;
      float dx = Pb[j * 4 + 0] - cx, dy = Pb[j * 4 + 1] - cy;
      float dz = Pb[j * 4 + 2] - cz, dw = Pb[j * 4 + 3] - cw;
      float d = dx * dx + dy * dy + dz * dz + dw * dw;
      bool cond = d <= r2;
      unsigned mask = (unsigned)__ballot(cond);
      int pos = __popc(mask & ((1u << lane) - 1u));
      if (cond && (total + pos) < nsample) idx_s[wv][total + pos] = j;
      total += __popc(mask);
    }
    if (total > nsample) total = nsample;
  }
  __syncthreads();
  if (valid) {
    int first = (total > 0) ? idx_s[wv][0] : 0;
    for (int t = lane; t < nsample; t += 32)
      if (t >= total) idx_s[wv][t] = first;
  }
  __syncthreads();
  if (valid) {
    for (int t = lane; t < nsample; t += 32) {
      int j = idx_s[wv][t];
      size_t row = (size_t)gw * nsample + t;
      half_t* o = out + row * outCpad;
      o[0] = (half_t)(Pb[j * 4 + 0] - cx);
      o[1] = (half_t)(Pb[j * 4 + 1] - cy);
      o[2] = (half_t)(Pb[j * 4 + 2] - cz);
      o[3] = (half_t)(Pb[j * 4 + 3] - cw);
      const half_t* f = feat ? feat + ((size_t)b * N + j) * featC : nullptr;
      for (int c = 0; c < featC; ++c) o[4 + c] = f[c];
      for (int c = 4 + featC; c < outCpad; ++c) o[c] = (half_t)0.f;
    }
  }
}

// ---------------------------------------------------------------- WMMA GEMM
// Y = A @ Wh^T + bias. A: (M x Kpad) f16 raw; Wh: (>=ceil64(Nout) x Kpad) f16.
// Block = 256 threads (8 wave32), output tile 128(M) x 64(N), K-chunks of 32.
// Tiles staged with GLOBAL_LOAD_ASYNC_TO_LDS_B128; each wave reuses one
// A-fragment against 4 B-fragments -> 4 v_wmma per wave per chunk.
// Per-channel sum/sumsq accumulated into stats[2*Nout] for batch-norm.
__global__ void wmma_gemm_kernel(const half_t* __restrict__ A, int Kpad,
                                 const half_t* __restrict__ Wh,
                                 const float* __restrict__ bias,
                                 int M, int Nout,
                                 half_t* __restrict__ Y, float* __restrict__ stats) {
  __shared__ half_t As[128 * 32];
  __shared__ half_t Ws[64 * 32];
  __shared__ float  statS[128];

  const int tid = threadIdx.x;
  const int lane = tid & 31;
  const int wave = tid >> 5;
  const int row0 = blockIdx.x * 128;
  const int col0 = blockIdx.y * 64;

  for (int i = tid; i < 128; i += 256) statS[i] = 0.f;

  v8f acc0 = {}, acc1 = {}, acc2 = {}, acc3 = {};
  const int lrow = lane & 15;
  const int half_id = lane >> 4;

  uint32_t asbase = (uint32_t)(uintptr_t)&As[0];
  uint32_t wsbase = (uint32_t)(uintptr_t)&Ws[0];

  for (int kc = 0; kc < Kpad; kc += 32) {
    // A tile 128x32 f16 = 512 x 16B chunks; 2 per thread
    {
      int c0 = tid;
      int r = c0 >> 2, cc = c0 & 3;
      async_copy16(asbase + c0 * 16, A + (size_t)(row0 + r) * Kpad + kc + cc * 8);
      int c1 = tid + 256;
      r = c1 >> 2; cc = c1 & 3;
      async_copy16(asbase + c1 * 16, A + (size_t)(row0 + r) * Kpad + kc + cc * 8);
    }
    // W tile 64x32 f16 = 256 x 16B chunks; 1 per thread
    {
      int r = tid >> 2, cc = tid & 3;
      async_copy16(wsbase + tid * 16, Wh + (size_t)(col0 + r) * Kpad + kc + cc * 8);
    }
    wait_async();
    __syncthreads();

    v16h afrag, b0, b1, b2, b3;
#pragma unroll
    for (int e = 0; e < 16; ++e) {
      int ka = e + 16 * half_id;                       // A: row per lane, K linear
      int kb = e + 16 * half_id;                       // B: column per lane, K linear
      afrag[e] = As[(wave * 16 + lrow) * 32 + ka];
      b0[e] = Ws[(0 * 16 + lrow) * 32 + kb];
      b1[e] = Ws[(1 * 16 + lrow) * 32 + kb];
      b2[e] = Ws[(2 * 16 + lrow) * 32 + kb];
      b3[e] = Ws[(3 * 16 + lrow) * 32 + kb];
    }
    acc0 = __builtin_amdgcn_wmma_f32_16x16x32_f16(false, afrag, false, b0, (short)0, acc0, false, false);
    acc1 = __builtin_amdgcn_wmma_f32_16x16x32_f16(false, afrag, false, b1, (short)0, acc1, false, false);
    acc2 = __builtin_amdgcn_wmma_f32_16x16x32_f16(false, afrag, false, b2, (short)0, acc2, false, false);
    acc3 = __builtin_amdgcn_wmma_f32_16x16x32_f16(false, afrag, false, b3, (short)0, acc3, false, false);
    __syncthreads();
  }

  v8f accs[4] = {acc0, acc1, acc2, acc3};
#pragma unroll
  for (int nt = 0; nt < 4; ++nt) {
    int n = col0 + nt * 16 + lrow;
    float bv = (n < Nout) ? bias[n] : 0.f;
    float s = 0.f, sq = 0.f;
    if (n < Nout) {
#pragma unroll
      for (int v = 0; v < 8; ++v) {
        int m = row0 + wave * 16 + half_id * 8 + v;
        if (m < M) {
          float val = accs[nt][v] + bv;
          Y[(size_t)m * Nout + n] = (half_t)val;
          s += val; sq += val * val;
        }
      }
    }
    if (stats) {
      s  += __shfl_xor(s, 16, 32);
      sq += __shfl_xor(sq, 16, 32);
      if (half_id == 0) {
        atomicAdd(&statS[nt * 16 + lrow], s);
        atomicAdd(&statS[64 + nt * 16 + lrow], sq);
      }
    }
  }
  if (stats) {
    __syncthreads();
    if (tid < 64) {
      int n2 = col0 + tid;
      if (n2 < Nout) {
        atomicAdd(&stats[n2],        statS[tid]);
        atomicAdd(&stats[Nout + n2], statS[64 + tid]);
      }
    }
  }
}

__global__ void bn_finalize_kernel(const float* __restrict__ stats,
                                   const float* __restrict__ gamma, const float* __restrict__ beta,
                                   float invcount, int C,
                                   float* __restrict__ scale, float* __restrict__ shift) {
  int c = blockIdx.x * blockDim.x + threadIdx.x;
  if (c >= C) return;
  float mean = stats[c] * invcount;
  float var  = stats[C + c] * invcount - mean * mean;
  float sc = gamma[c] * rsqrtf(var + 1e-5f);
  scale[c] = sc;
  shift[c] = beta[c] - mean * sc;
}

// plain max over already-activated f16 rows
__global__ void maxpool_kernel(const half_t* __restrict__ Y, int groups, int nsample, int C,
                               half_t* __restrict__ out) {
  int i = blockIdx.x * blockDim.x + threadIdx.x;
  if (i >= groups * C) return;
  int g = i / C, c = i % C;
  const half_t* p = Y + (size_t)g * nsample * C + c;
  float m = -1e30f;
  for (int j = 0; j < nsample; ++j) {
    float v = (float)p[(size_t)j * C];
    if (v > m) m = v;
  }
  out[i] = (half_t)m;
}

// grouped3[r, 0:4]=l2_xyz, [4:260]=feat2, [260:288]=0  (stride 288)
__global__ void concat_sa3_kernel(const float* __restrict__ nx2, const half_t* __restrict__ feat2,
                                  half_t* __restrict__ out, int rows) {
  int i = blockIdx.x * blockDim.x + threadIdx.x;
  if (i >= rows * 288) return;
  int r = i / 288, c = i % 288;
  float v = 0.f;
  if (c < 4) v = nx2[r * 4 + c];
  else if (c < 260) v = (float)feat2[(size_t)r * 256 + (c - 4)];
  out[i] = (half_t)v;
}

__global__ void softmax_kernel(const half_t* __restrict__ logits, float* __restrict__ out, int C) {
  __shared__ float red[64];
  int r = blockIdx.x, t = threadIdx.x;
  float v = (t < C) ? (float)logits[(size_t)r * C + t] : -1e30f;
  red[t] = v;
  __syncthreads();
  for (int s = 32; s > 0; s >>= 1) { if (t < s) red[t] = fmaxf(red[t], red[t + s]); __syncthreads(); }
  float mx = red[0];
  __syncthreads();
  float e = (t < C) ? expf(v - mx) : 0.f;
  red[t] = e;
  __syncthreads();
  for (int s = 32; s > 0; s >>= 1) { if (t < s) red[t] += red[t + s]; __syncthreads(); }
  if (t < C) out[(size_t)r * C + t] = e / red[0];
}

// ---------------------------------------------------------------- driver

extern "C" void kernel_launch(void* const* d_in, const int* in_sizes, int n_in,
                              void* d_out, int out_size, void* d_ws, size_t ws_size,
                              hipStream_t stream) {
  (void)in_sizes; (void)n_in; (void)out_size; (void)ws_size;
  const int B = 32, N = 4096;
  const int S1 = 512, NS1 = 32;
  const int S2 = 128, NS2 = 64;

  auto P = [&](int i) { return (const float*)d_in[i]; };
  // input order: xyz; sa1(3x w,b,g,be)@1; sa2@13; sa3@25; fc1 w/b@37; bn1@39; fc2@41; bn2@43; fc3@45

  char* ws = (char*)d_ws;
  size_t off = 0;
  auto alloc = [&](size_t bytes) -> void* {
    void* p = ws + off;
    off += (bytes + 255) & ~(size_t)255;
    return p;
  };

  float*  pts      = (float*)alloc((size_t)B * N * 4 * 4);
  int*    idx1     = (int*)alloc((size_t)B * S1 * 4);
  float*  nx1      = (float*)alloc((size_t)B * S1 * 4 * 4);
  int*    idx2     = (int*)alloc((size_t)B * S2 * 4);
  float*  nx2      = (float*)alloc((size_t)B * S2 * 4 * 4);
  half_t* grouped1 = (half_t*)alloc((size_t)B * S1 * NS1 * 32 * 2);   // Kpad=32
  half_t* grouped2 = (half_t*)alloc((size_t)B * S2 * NS2 * 160 * 2);  // Kpad=160
  half_t* grouped3 = (half_t*)alloc((size_t)B * S2 * 288 * 2);        // Kpad=288
  half_t* feat1    = (half_t*)alloc((size_t)B * S1 * 128 * 2);
  half_t* feat2    = (half_t*)alloc((size_t)B * S2 * 256 * 2);
  half_t* feat3    = (half_t*)alloc((size_t)128 * 1024 * 2);          // rows padded to 128
  half_t* Wh       = (half_t*)alloc((size_t)1024 * 1024 * 2 + 65536); // padded f16 weights
  float*  stats    = (float*)alloc(2 * 1024 * 4);
  float*  sc0 = (float*)alloc(1024 * 4);
  float*  sh0 = (float*)alloc(1024 * 4);
  float*  sc1 = (float*)alloc(1024 * 4);
  float*  sh1 = (float*)alloc(1024 * 4);
  half_t* logits = (half_t*)alloc((size_t)B * 48 * 2);
  half_t* bufY0  = (half_t*)alloc((size_t)524288 * 128 * 2);
  half_t* bufY1  = (half_t*)alloc((size_t)524288 * 128 * 2);

  // one MLP layer: convert weights, GEMM(+stats), fold BN, apply BN+ReLU in place
  auto layer = [&](const half_t* A, int Kpad, int M, int K, int widx, int Nout,
                   half_t* Yp, float* scp, float* shp, bool do_bn) {
    {
      int tot = Nout * Kpad;
      convert_w_kernel<<<(tot + 255) / 256, 256, 0, stream>>>(P(widx), Nout, K, Kpad, Wh);
    }
    if (do_bn) {
      int n2 = 2 * Nout;
      zero_kernel<<<(n2 + 255) / 256, 256, 0, stream>>>(stats, n2);
    }
    dim3 g((M + 127) / 128, (Nout + 63) / 64);
    wmma_gemm_kernel<<<g, 256, 0, stream>>>(A, Kpad, Wh, P(widx + 1), M, Nout, Yp,
                                            do_bn ? stats : nullptr);
    if (do_bn) {
      bn_finalize_kernel<<<(Nout + 63) / 64, 64, 0, stream>>>(stats, P(widx + 2), P(widx + 3),
                                                              1.0f / (float)M, Nout, scp, shp);
      int tot = M * Nout;
      bnrelu_kernel<<<(tot + 255) / 256, 256, 0, stream>>>(Yp, scp, shp, M, Nout);
    }
  };

  // pts = swapaxes(xyz,1,2)
  {
    int tot = B * 4 * N;
    transpose_xyz_kernel<<<(tot + 255) / 256, 256, 0, stream>>>((const float*)d_in[0], pts, B, 4, N);
  }

  // ---- SA1: npoint=512, r=0.2, nsample=32, MLP 4->64->64->128
  fps_kernel<<<B, 256, 0, stream>>>(pts, N, S1, idx1);
  {
    int tot = B * S1 * 4;
    gather_xyz_kernel<<<(tot + 255) / 256, 256, 0, stream>>>(pts, idx1, N, S1, tot, nx1);
  }
  ballquery_group_kernel<<<(B * S1) / 8, 256, 0, stream>>>(pts, nullptr, nx1, B, N, S1, NS1,
                                                          0.04f, 0, 32, grouped1);
  const int M1 = B * S1 * NS1;
  layer(grouped1, 32, M1, 4,  1, 64,  bufY0, sc0, sh0, true);
  layer(bufY0,    64, M1, 64, 5, 64,  bufY1, sc0, sh0, true);
  layer(bufY1,    64, M1, 64, 9, 128, bufY0, sc0, sh0, true);
  {
    int tot = B * S1 * 128;
    maxpool_kernel<<<(tot + 255) / 256, 256, 0, stream>>>(bufY0, B * S1, NS1, 128, feat1);
  }

  // ---- SA2: npoint=128, r=0.4, nsample=64, MLP 132->128->128->256
  fps_kernel<<<B, 256, 0, stream>>>(nx1, S1, S2, idx2);
  {
    int tot = B * S2 * 4;
    gather_xyz_kernel<<<(tot + 255) / 256, 256, 0, stream>>>(nx1, idx2, S1, S2, tot, nx2);
  }
  ballquery_group_kernel<<<(B * S2) / 8, 256, 0, stream>>>(nx1, feat1, nx2, B, S1, S2, NS2,
                                                          0.16f, 128, 160, grouped2);
  const int M2 = B * S2 * NS2;
  layer(grouped2, 160, M2, 132, 13, 128, bufY0, sc0, sh0, true);
  layer(bufY0,    128, M2, 128, 17, 128, bufY1, sc0, sh0, true);
  layer(bufY1,    128, M2, 128, 21, 256, bufY0, sc0, sh0, true);
  {
    int tot = B * S2 * 256;
    maxpool_kernel<<<(tot + 255) / 256, 256, 0, stream>>>(bufY0, B * S2, NS2, 256, feat2);
  }

  // ---- SA3 (group-all): MLP 260->256->512->1024, max over 128 points
  {
    int tot = B * S2 * 288;
    concat_sa3_kernel<<<(tot + 255) / 256, 256, 0, stream>>>(nx2, feat2, grouped3, B * S2);
  }
  const int M3 = B * S2;
  layer(grouped3, 288, M3, 260, 25, 256,  bufY0, sc0, sh0, true);
  layer(bufY0,    256, M3, 256, 29, 512,  bufY1, sc0, sh0, true);
  layer(bufY1,    512, M3, 512, 33, 1024, bufY0, sc0, sh0, true);
  {
    int tot = B * 1024;
    maxpool_kernel<<<(tot + 255) / 256, 256, 0, stream>>>(bufY0, B, S2, 1024, feat3);
  }

  // ---- FC head: 1024->512 (BN,relu) -> 256 (BN,relu) -> 40 -> softmax
  layer(feat3, 1024, B, 1024, 37, 512, bufY1, sc0, sh0, true);  // uses bn1 params @39/40
  layer(bufY1, 512,  B, 512,  41, 256, bufY0, sc1, sh1, true);  // uses bn2 params @43/44
  layer(bufY0, 256,  B, 256,  45, 40,  logits, nullptr, nullptr, false);
  softmax_kernel<<<B, 64, 0, stream>>>(logits, (float*)d_out, 40);
}
